// CKSAAP_39204461478229
// MI455X (gfx1250) — matrix operand order repaired and checked by reading, not compiled, not run on hardware
//
#include <hip/hip_runtime.h>

typedef __attribute__((ext_vector_type(16))) _Float16       v16h;
typedef __attribute__((ext_vector_type(2)))  _Float16       h2;
typedef __attribute__((ext_vector_type(8)))  float          v8f;
typedef __attribute__((ext_vector_type(8)))  unsigned short u16x8;

#define NUM_AA    20
#define NUM_PAIRS 400           // 20*20
#define B_DIM     32
#define L_DIM     4096
#define D_DIM     64
#define CHUNK     32            // K per v_wmma_f32_16x16x32_f16
#define M_TILES   25            // 400/16 exactly -- no padding, no epilogue guard
#define TILES_PER_WAVE 13       // msets cover tiles 0-12 and 12-24 (tile 12 shared)
#define PE_STRIDE 40            // halves per d-row: 32 + 8 pad = 80 B (16B multiple)

__launch_bounds__(256, 1)
__global__ void cksaap_onehot_wmma(const int* __restrict__ seq,
                                   const float* __restrict__ emb,
                                   float* __restrict__ out,
                                   int K1) {
    const int wg = blockIdx.x;          // = b*K1 + t
    const int t  = wg % K1;
    const int b  = wg / K1;
    const int n  = L_DIM - t - 1;

    __shared__ __align__(16) _Float16       sPE[2][D_DIM][PE_STRIDE];   // 10 KB
    __shared__ __align__(16) _Float16       sFrag[2][M_TILES][32][16];  // 50 KB
    __shared__ __align__(16) unsigned short sIdx[2][CHUNK];             // 128 B

    const int tid  = threadIdx.x;
    const int lane = tid & 31;
    const int wave = tid >> 5;
    const int ni   = wave & 3;          // N-tile (d block) for this wave
    const int mset = wave >> 2;         // tile window: 0 -> 0..12, 1 -> 12..24
    const int rowq = lane & 15;
    const int hi   = lane >> 4;

    v8f acc[TILES_PER_WAVE];
#pragma unroll
    for (int i = 0; i < TILES_PER_WAVE; ++i) {
        v8f z = {0.f, 0.f, 0.f, 0.f, 0.f, 0.f, 0.f, 0.f};
        acc[i] = z;
    }

    const int  d_st  = tid & 63;        // staging: 64 d-lanes x 4 k-quarters
    const int  q_st  = tid >> 6;        // wave-uniform
    const long base0 = (long)b * L_DIM * D_DIM;
    const int  nchunks = (n + CHUNK - 1) / CHUNK;   // 128

    // ---- stage chunk c into buffer bf (branch-free: clamp + mask-scale) ----
    auto stage = [&](int c, int bf) {
        const int  kq = c * CHUNK + q_st * 8;       // wave-uniform first k
        const long eb = base0 + d_st;
#pragma unroll
        for (int r2 = 0; r2 < 4; ++r2) {
            const int   i0 = kq + 2 * r2;
            const int   i1 = i0 + 1;
            const int   c0 = (i0 < n - 1) ? i0 : n - 1;     // clamp (uniform)
            const int   c1 = (i1 < n - 1) ? i1 : n - 1;
            const float s0 = (i0 < n) ? 0.5f : 0.0f;        // zero masked tail
            const float s1 = (i1 < n) ? 0.5f : 0.0f;
            const float v0 = (emb[eb + (long)c0 * D_DIM] +
                              emb[eb + (long)(c0 + t + 1) * D_DIM]) * s0;
            const float v1 = (emb[eb + (long)c1 * D_DIM] +
                              emb[eb + (long)(c1 + t + 1) * D_DIM]) * s1;
            h2 pk;
            pk.x = (_Float16)v0;
            pk.y = (_Float16)v1;
            *(h2*)&sPE[bf][d_st][q_st * 8 + 2 * r2] = pk;   // ds_store_b32
        }
        if (tid < CHUNK) {
            const int i  = c * CHUNK + tid;
            const int ci = (i < n - 1) ? i : n - 1;         // clamp, stays in-bounds
            const int pr = seq[b * L_DIM + ci] * NUM_AA +
                           seq[b * L_DIM + ci + t + 1];
            sIdx[bf][tid] = (unsigned short)((i < n) ? pr : 1024);  // 1024: no row
        }
    };

    // ---- build each unique one-hot A tile exactly once (static assignment) ----
    // 16-bit A layout (contiguous, matches B): lanes 0-15 K=0..15, lanes 16-31 K=16..31
    auto produce = [&](int bf) {
        const u16x8 ivlo = *(const u16x8*)&sIdx[bf][hi * 16];
        const u16x8 ivhi = *(const u16x8*)&sIdx[bf][hi * 16 + 8];
        auto build = [&](int p) {
            const unsigned short myrow = (unsigned short)(p * 16 + rowq);
            v16h av;
#pragma unroll
            for (int j = 0; j < 8; ++j) {
                av[j]     = (ivlo[j] == myrow) ? (_Float16)1.0f : (_Float16)0.0f;
                av[j + 8] = (ivhi[j] == myrow) ? (_Float16)1.0f : (_Float16)0.0f;
            }
            *(v16h*)&sFrag[bf][p][lane][0] = av;            // 2x ds_store_b128
        };
#pragma unroll
        for (int m = 0; m < 3; ++m) build(wave + 8 * m);    // tiles 0..23
        if (wave == 0) build(24);                           // uniform scalar branch
    };

    // ---- consume buffer bf: 13 software-pipelined WMMAs ----
    auto consume = [&](int bf) {
        const int  d  = ni * 16 + rowq;
        const v16h bv = *(const v16h*)&sPE[bf][d][hi * 16]; // 2x ds_load_b128
        const _Float16* fb = &sFrag[bf][mset * 12][lane][0];
        v16h a0 = *(const v16h*)fb;                         // preload tile 0
#pragma unroll
        for (int mm = 0; mm < TILES_PER_WAVE; ++mm) {
            v16h a1 = a0;
            if (mm + 1 < TILES_PER_WAVE)                    // load next while MAC runs
                a1 = *(const v16h*)(fb + (long)(mm + 1) * 32 * 16);
            acc[mm] = __builtin_amdgcn_wmma_f32_16x16x32_f16(
                false, a0, false, bv, (short)0, acc[mm], false, false);
            a0 = a1;
        }
    };

    // prologue: fill buffer 0
    stage(0, 0);
    __syncthreads();
    produce(0);
    __syncthreads();

    for (int c = 0; c < nchunks; ++c) {
        const int buf = c & 1, nb = buf ^ 1;
        if (c + 1 < nchunks) stage(c + 1, nb);   // overlaps with consume(buf)
        consume(buf);
        __syncthreads();                         // sIdx[nb] visible; frag[buf] reads done
        if (c + 1 < nchunks) produce(nb);
        __syncthreads();                         // frag[nb] + sPE[nb] visible
    }

    // ---- epilogue: scale by 1/n, write [B, k+1, 20, 20, D] (no guards) ----
    const float inv_n = 1.0f / (float)n;
    float* outp = out + (long)(b * K1 + t) * NUM_PAIRS * D_DIM;
    const int d = ni * 16 + rowq;               // C layout: N = lane % 16
#pragma unroll
    for (int mm = 0; mm < TILES_PER_WAVE; ++mm) {
        const int rowbase = (mset * 12 + mm) * 16;
#pragma unroll
        for (int v = 0; v < 8; ++v) {
            const int pair = rowbase + v + hi * 8;  // C layout: M = vgpr (+8 hi lanes)
            outp[(long)pair * D_DIM + d] = acc[mm][v] * inv_n;
        }
    }
}

extern "C" void kernel_launch(void* const* d_in, const int* in_sizes, int n_in,
                              void* d_out, int out_size, void* d_ws, size_t ws_size,
                              hipStream_t stream) {
    (void)d_ws; (void)ws_size; (void)n_in;
    const int*   seq  = (const int*)d_in[0];
    const float* emb  = (const float*)d_in[1];
    float*       outp = (float*)d_out;

    const int D  = in_sizes[1] / in_sizes[0];           // 64
    const int K1 = out_size / (B_DIM * NUM_PAIRS * D);  // k+1 = 8

    dim3 grid(B_DIM * K1);   // 256 workgroups: one per (batch, gap)
    dim3 block(256);         // 8 wave32
    hipLaunchKernelGGL(cksaap_onehot_wmma, grid, block, 0, stream,
                       seq, emb, outp, K1);
}
